// steerable_u_net_79843442032658
// MI455X (gfx1250) — matrix operand order repaired and checked by reading, not compiled, not run on hardware
//
#include <hip/hip_runtime.h>
#include <cstddef>
#include <cstdint>

// ---------------------------------------------------------------------------
// Types
// ---------------------------------------------------------------------------
typedef __bf16 bf16_t;
typedef bf16_t v16bf __attribute__((ext_vector_type(16)));
typedef bf16_t v8bf  __attribute__((ext_vector_type(8)));
typedef float  v8f   __attribute__((ext_vector_type(8)));

// ---------------------------------------------------------------------------
// C4 rotation source index: (si,sj) such that rot^g(A)[i][j] == A[si][sj]
// rot90 (numpy, axes=(-2,-1)): B[i][j] = A[j][K-1-i]
// ---------------------------------------------------------------------------
__device__ __forceinline__ void rotsrc(int g, int i, int j, int& si, int& sj) {
    si = i; sj = j;
    for (int t = 0; t < g; ++t) { int ni = sj, nj = 2 - si; si = ni; sj = nj; }
}

// ---------------------------------------------------------------------------
// Weight expansion kernels: base f32 params -> bf16 [Cout][9][Cin] (GEMM layout)
// ---------------------------------------------------------------------------
__global__ void expand_lift_k(const float* __restrict__ w, bf16_t* __restrict__ out,
                              int Of, int Ic) {
    int i = blockIdx.x * blockDim.x + threadIdx.x;
    int total = Of * 4 * 9 * 32;
    if (i >= total) return;
    int ci = i & 31;
    int t  = (i >> 5) % 9;
    int oc = i / (32 * 9);
    int o = oc >> 2, g = oc & 3;
    bf16_t v = (bf16_t)0.0f;
    if (ci < Ic) {
        int si, sj; rotsrc(g, t / 3, t % 3, si, sj);
        v = (bf16_t)w[((o * Ic + ci) * 3 + si) * 3 + sj];
    }
    out[i] = v;
}

// reg: w [Of, If, 4, 3, 3] -> out [Of*4][9][If*4]
// expanded[of*4+g][if*4+h] = rot_g(w[of][if][(h-g)%4])
__global__ void expand_reg_k(const float* __restrict__ w, bf16_t* __restrict__ out,
                             int Of, int If) {
    int Cin = If * 4;
    int i = blockIdx.x * blockDim.x + threadIdx.x;
    int total = Of * 4 * 9 * Cin;
    if (i >= total) return;
    int cin = i % Cin;
    int t   = (i / Cin) % 9;
    int oc  = i / (Cin * 9);
    int of = oc >> 2, g = oc & 3;
    int if_ = cin >> 2, h = cin & 3;
    int hh = (h - g) & 3;
    int si, sj; rotsrc(g, t / 3, t % 3, si, sj);
    out[i] = (bf16_t)w[((((size_t)of * If + if_) * 4 + hh) * 3 + si) * 3 + sj];
}

// proj: w [Oc, If, 3, 3] -> out [Oc][9][If*4]; in channel if*4+g uses rot_g
__global__ void expand_proj_k(const float* __restrict__ w, bf16_t* __restrict__ out,
                              int Oc, int If) {
    int Cin = If * 4;
    int i = blockIdx.x * blockDim.x + threadIdx.x;
    int total = Oc * 9 * Cin;
    if (i >= total) return;
    int cin = i % Cin;
    int t   = (i / Cin) % 9;
    int oc  = i / (Cin * 9);
    int if_ = cin >> 2, g = cin & 3;
    int si, sj; rotsrc(g, t / 3, t % 3, si, sj);
    out[i] = (bf16_t)w[(((size_t)oc * If + if_) * 3 + si) * 3 + sj];
}

__global__ void expand_bias_k(const float* __restrict__ b, float* __restrict__ out,
                              int Cout, int rep) {
    int i = blockIdx.x * blockDim.x + threadIdx.x;
    if (i < Cout) out[i] = b[i / rep];
}

// ---------------------------------------------------------------------------
// Input conversion: x [B,C,H,W] f32 NCHW -> bf16 NHWC with channels padded to CP
// ---------------------------------------------------------------------------
__global__ void cvtin_k(const float* __restrict__ x, bf16_t* __restrict__ out,
                        int B, int C, int H, int W, int CP) {
    size_t i = (size_t)blockIdx.x * blockDim.x + threadIdx.x;
    size_t total = (size_t)B * H * W * CP;
    if (i >= total) return;
    int c = (int)(i % CP);
    size_t p = i / CP;
    int xx = (int)(p % W); p /= W;
    int y  = (int)(p % H);
    int b  = (int)(p / H);
    float v = 0.0f;
    if (c < C) v = x[(((size_t)b * C + c) * H + y) * W + xx];
    out[i] = (bf16_t)v;
}

// ---------------------------------------------------------------------------
// Implicit-GEMM 3x3 conv (pad=1) + bias + ReLU via v_wmma_f32_16x16x32_bf16,
// with double-buffered async global->LDS halo staging (ASYNCcnt path).
// in:  NHWC bf16, C = CIN (multiple of 32, compile-time)
// w:   [Cout][9][CIN] bf16 (tap-major, channel contiguous)
// out: NHWC bf16, C = Cout (multiple of 16)
// Block: 256 thr = 8 waves = 4 cout-tiles x 2 pixel-tiles (64 cout x 32 px).
// ---------------------------------------------------------------------------
template <int CIN>
__global__ __launch_bounds__(256)
void conv3x3_wmma_k(const bf16_t* __restrict__ in, const bf16_t* __restrict__ wexp,
                    const float* __restrict__ bias, bf16_t* __restrict__ out,
                    int B, int H, int W, int Cout) {
    // two halo buffers: rows(-1..1) x cols(x0-1..x0+32) x 32ch
    __shared__ __align__(64) bf16_t lds[2][3][34][32];

    const int tiles_x = (W + 31) >> 5;
    int bx  = blockIdx.x % tiles_x;
    int rem = blockIdx.x / tiles_x;
    int y   = rem % H;
    int b   = rem / H;
    int x0  = bx << 5;

    const int tid   = threadIdx.x;
    const int lane  = tid & 31;
    const int wave  = tid >> 5;
    const int m     = lane & 15;   // A row (cout) / B col (pixel)
    const int hi    = lane >> 4;   // K-half selector
    const int waveC = wave >> 1;
    const int waveP = wave & 1;

    int cbase = blockIdx.y * 64 + waveC * 16;
    if (cbase > Cout - 16) cbase = Cout - 16;   // clamp (duplicate tiles write same data)

    const uintptr_t lds0 = (uintptr_t)&lds[0][0][0][0];   // LDS allocation starts at 0

    // ---- one-time zero-fill of out-of-bounds halo slots (both buffers) ----
    for (int t = tid; t < 408; t += 256) {                // 102 slots * 4 x 16B
        int slot = t >> 2, part = t & 3;
        int row = slot / 34, col = slot - row * 34;
        int gy = y + row - 1;
        int gx = x0 + col - 1;
        if (!(gy >= 0 && gy < H && gx >= 0 && gx < W)) {
            uint4 z = make_uint4(0u, 0u, 0u, 0u);
            *(uint4*)(&lds[0][row][col][part * 8]) = z;
            *(uint4*)(&lds[1][row][col][part * 8]) = z;
        }
    }

    // ---- async stage of one 32-channel chunk into buffer `buf` ----
    auto stage_async = [&](int buf, int ci0) {
        for (int t = tid; t < 408; t += 256) {
            int slot = t >> 2, part = t & 3;
            int row = slot / 34, col = slot - row * 34;
            int gy = y + row - 1;
            int gx = x0 + col - 1;
            if (gy >= 0 && gy < H && gx >= 0 && gx < W) {
                uint64_t ga = (uint64_t)(uintptr_t)(in
                    + (((size_t)b * H + gy) * W + gx) * CIN + ci0 + part * 8);
                unsigned lo = (unsigned)((uintptr_t)&lds[buf][row][col][part * 8] - lds0);
                asm volatile("global_load_async_to_lds_b128 %0, %1, off"
                             :: "v"(lo), "v"(ga) : "memory");
            }
        }
    };

    v8f acc = {0.f, 0.f, 0.f, 0.f, 0.f, 0.f, 0.f, 0.f};

    constexpr int NC = CIN / 32;
    stage_async(0, 0);
    asm volatile("s_wait_asynccnt 0x0" ::: "memory");
    __syncthreads();

    #pragma unroll 1
    for (int c = 0; c < NC; ++c) {
        int cur = c & 1;
        if (c + 1 < NC) stage_async(cur ^ 1, (c + 1) * 32);   // prefetch next chunk

        // A row base for this lane: W[cbase+m][tap][c*32 + hi*16 .. +15]
        const bf16_t* wrow = wexp + (size_t)(cbase + m) * 9 * CIN + c * 32 + hi * 16;

        #pragma unroll
        for (int tap = 0; tap < 9; ++tap) {
            const int dy = tap / 3, dx = tap % 3;
            // A fragment: contiguous K order (lane<16: K=0..15; lane>=16: K=16..31)
            v16bf aF = *(const v16bf*)(wrow + tap * CIN);
            // B fragment: contiguous K order, same as A (lane<16: K=0..15; lane>=16: K=16..31)
            int col = waveP * 16 + m + dx;
            v16bf bF = *(const v16bf*)(&lds[cur][dy][col][hi * 16]);
            acc = __builtin_amdgcn_wmma_f32_16x16x32_bf16(
                false, aF, false, bF, (short)0, acc, false, false);
        }

        asm volatile("s_wait_asynccnt 0x0" ::: "memory");  // prefetch done
        __syncthreads();                                   // all reads of cur done
    }

    // ---- bias + ReLU + pack 8 x bf16, one 16B store per lane ----
    int gx = x0 + waveP * 16 + m;
    if (gx < W) {
        v8f bv = *(const v8f*)(bias + cbase + hi * 8);
        v8bf o;
        #pragma unroll
        for (int r = 0; r < 8; ++r) {
            float v = acc[r] + bv[r];
            o[r] = (bf16_t)(v > 0.f ? v : 0.f);
        }
        *(v8bf*)(out + (((size_t)b * H + y) * W + gx) * Cout + cbase + hi * 8) = o;
    }
}

// ---------------------------------------------------------------------------
// 2x2 avg pool, NHWC bf16
// ---------------------------------------------------------------------------
__global__ void pool2_k(const bf16_t* __restrict__ in, bf16_t* __restrict__ out,
                        int B, int C, int Ho, int Wo) {
    size_t i = (size_t)blockIdx.x * blockDim.x + threadIdx.x;
    size_t total = (size_t)B * Ho * Wo * C;
    if (i >= total) return;
    int c = (int)(i % C);
    size_t p = i / C;
    int x = (int)(p % Wo); p /= Wo;
    int y = (int)(p % Ho);
    int b = (int)(p / Ho);
    int Wi = Wo * 2;
    const bf16_t* base = in + (((size_t)b * (Ho * 2) + 2 * y) * Wi + 2 * x) * C + c;
    float s = (float)base[0] + (float)base[C]
            + (float)base[(size_t)Wi * C] + (float)base[(size_t)Wi * C + C];
    out[i] = (bf16_t)(0.25f * s);
}

// ---------------------------------------------------------------------------
// Bilinear x2 upsample (half-pixel, align_corners=False), NHWC bf16,
// writes channels [0,C) of a concat buffer with per-pixel stride CT.
// ---------------------------------------------------------------------------
__global__ void up2_k(const bf16_t* __restrict__ in, bf16_t* __restrict__ out,
                      int B, int C, int Hi, int Wi, int CT) {
    const int Ho = Hi * 2, Wo = Wi * 2;
    size_t i = (size_t)blockIdx.x * blockDim.x + threadIdx.x;
    size_t total = (size_t)B * Ho * Wo * C;
    if (i >= total) return;
    int c = (int)(i % C);
    size_t p = i / C;
    int x = (int)(p % Wo); p /= Wo;
    int y = (int)(p % Ho);
    int b = (int)(p / Ho);
    int y0 = (y - 1) >> 1;  float fy = (y & 1) ? 0.25f : 0.75f;
    int x0 = (x - 1) >> 1;  float fx = (x & 1) ? 0.25f : 0.75f;
    int ya = y0 < 0 ? 0 : y0;             int yb = (y0 + 1 > Hi - 1) ? Hi - 1 : y0 + 1;
    int xa = x0 < 0 ? 0 : x0;             int xb = (x0 + 1 > Wi - 1) ? Wi - 1 : x0 + 1;
    const bf16_t* src = in + ((size_t)b * Hi) * Wi * C + c;
    float vaa = (float)src[((size_t)ya * Wi + xa) * C];
    float vab = (float)src[((size_t)ya * Wi + xb) * C];
    float vba = (float)src[((size_t)yb * Wi + xa) * C];
    float vbb = (float)src[((size_t)yb * Wi + xb) * C];
    float v = (1.f - fy) * ((1.f - fx) * vaa + fx * vab)
            +        fy  * ((1.f - fx) * vba + fx * vbb);
    out[(((size_t)b * Ho + y) * Wo + x) * CT + c] = (bf16_t)v;
}

// ---------------------------------------------------------------------------
// Skip copy into concat buffer at channel offset CO (stride CT), NHWC bf16
// ---------------------------------------------------------------------------
__global__ void copyoff_k(const bf16_t* __restrict__ in, bf16_t* __restrict__ out,
                          int B, int C, int H, int W, int CT, int CO) {
    size_t i = (size_t)blockIdx.x * blockDim.x + threadIdx.x;
    size_t total = (size_t)B * H * W * C;
    if (i >= total) return;
    int c = (int)(i % C);
    size_t pix = i / C;
    out[pix * CT + CO + c] = in[i];
}

// ---------------------------------------------------------------------------
// Final 1x1 conv (trivial->trivial), bf16 NHWC -> f32 NCHW (C=1), no ReLU
// ---------------------------------------------------------------------------
__global__ void out1x1_k(const bf16_t* __restrict__ in, const float* __restrict__ w,
                         const float* __restrict__ bb, float* __restrict__ out,
                         int B, int H, int W, int Cin) {
    size_t i = (size_t)blockIdx.x * blockDim.x + threadIdx.x;
    size_t total = (size_t)B * H * W;
    if (i >= total) return;
    const bf16_t* p = in + i * Cin;
    float acc = bb[0];
    for (int c = 0; c < Cin; ++c) acc += (float)p[c] * w[c];
    out[i] = acc;   // flat [b][0][y][x] == pixel order
}

// ---------------------------------------------------------------------------
// Host side
// ---------------------------------------------------------------------------
struct LayerDesc { int kind; int widx; int Of; int If; int Cout; int Cin; };
// kind: 0=lift, 1=reg, 2=proj

static void launch_conv(const bf16_t* in, const bf16_t* w, const float* bias,
                        bf16_t* out, int H, int W, int Cin, int Cout,
                        hipStream_t s) {
    dim3 grid((unsigned)(4 * H * ((W + 31) / 32)), (unsigned)((Cout + 63) / 64));
    switch (Cin) {
    case 32:   conv3x3_wmma_k<32>  <<<grid, 256, 0, s>>>(in, w, bias, out, 4, H, W, Cout); break;
    case 64:   conv3x3_wmma_k<64>  <<<grid, 256, 0, s>>>(in, w, bias, out, 4, H, W, Cout); break;
    case 128:  conv3x3_wmma_k<128> <<<grid, 256, 0, s>>>(in, w, bias, out, 4, H, W, Cout); break;
    case 192:  conv3x3_wmma_k<192> <<<grid, 256, 0, s>>>(in, w, bias, out, 4, H, W, Cout); break;
    case 256:  conv3x3_wmma_k<256> <<<grid, 256, 0, s>>>(in, w, bias, out, 4, H, W, Cout); break;
    case 384:  conv3x3_wmma_k<384> <<<grid, 256, 0, s>>>(in, w, bias, out, 4, H, W, Cout); break;
    case 512:  conv3x3_wmma_k<512> <<<grid, 256, 0, s>>>(in, w, bias, out, 4, H, W, Cout); break;
    case 768:  conv3x3_wmma_k<768> <<<grid, 256, 0, s>>>(in, w, bias, out, 4, H, W, Cout); break;
    case 1024: conv3x3_wmma_k<1024><<<grid, 256, 0, s>>>(in, w, bias, out, 4, H, W, Cout); break;
    case 1536: conv3x3_wmma_k<1536><<<grid, 256, 0, s>>>(in, w, bias, out, 4, H, W, Cout); break;
    default: break;
    }
}

extern "C" void kernel_launch(void* const* d_in, const int* in_sizes, int n_in,
                              void* d_out, int out_size, void* d_ws, size_t ws_size,
                              hipStream_t stream) {
    (void)in_sizes; (void)n_in; (void)out_size; (void)ws_size;

    static const LayerDesc L[18] = {
        {0,  1,  16,   2,   64,   32},  // d1c1 (lift, Cin padded to 32)
        {1,  3,  16,  16,   64,   64},  // d1c2
        {1,  5,  32,  16,  128,   64},  // d2c1
        {1,  7,  32,  32,  128,  128},  // d2c2
        {1,  9,  64,  32,  256,  128},  // d3c1
        {1, 11,  64,  64,  256,  256},  // d3c2
        {1, 13, 128,  64,  512,  256},  // d4c1
        {1, 15, 128, 128,  512,  512},  // d4c2
        {1, 17, 256, 128, 1024,  512},  // bn1
        {1, 19, 256, 256, 1024, 1024},  // bn2
        {1, 21, 128, 384,  512, 1536},  // u1c1
        {1, 23, 128, 128,  512,  512},  // u1c2
        {1, 25,  64, 192,  256,  768},  // u2c1
        {1, 27,  64,  64,  256,  256},  // u2c2
        {1, 29,  32,  96,  128,  384},  // u3c1
        {1, 31,  32,  32,  128,  128},  // u3c2
        {1, 33,  16,  48,   64,  192},  // u4c1
        {2, 35,  16,  16,   16,   64},  // u4c2 (proj)
    };

    // -------- bump allocator over d_ws --------
    char* base = (char*)d_ws;
    size_t off = 0;
    auto alloc = [&](size_t bytes) -> char* {
        char* p = base + off;
        off = (off + bytes + 255) & ~(size_t)255;
        return p;
    };

    bf16_t* Wp[18];
    float*  Bp[18];
    for (int l = 0; l < 18; ++l) {
        Wp[l] = (bf16_t*)alloc((size_t)L[l].Cout * 9 * L[l].Cin * sizeof(bf16_t));
        Bp[l] = (float*)alloc((size_t)L[l].Cout * sizeof(float));
    }

    const int Bn = 4;
    bf16_t* X  = (bf16_t*)alloc((size_t)Bn * 256 * 256 * 32  * sizeof(bf16_t));
    bf16_t* A  = (bf16_t*)alloc((size_t)Bn * 256 * 256 * 192 * sizeof(bf16_t));
    bf16_t* Bb = (bf16_t*)alloc((size_t)Bn * 256 * 256 * 192 * sizeof(bf16_t));
    bf16_t* S1 = (bf16_t*)alloc((size_t)Bn * 256 * 256 * 64  * sizeof(bf16_t));
    bf16_t* S2 = (bf16_t*)alloc((size_t)Bn * 128 * 128 * 128 * sizeof(bf16_t));
    bf16_t* S3 = (bf16_t*)alloc((size_t)Bn * 64  * 64  * 256 * sizeof(bf16_t));
    bf16_t* S4 = (bf16_t*)alloc((size_t)Bn * 32  * 32  * 512 * sizeof(bf16_t));

    // -------- expand weights + biases (bf16 GEMM layout) --------
    for (int l = 0; l < 18; ++l) {
        const float* w = (const float*)d_in[L[l].widx];
        const float* bsrc = (const float*)d_in[L[l].widx + 1];
        int total = L[l].Cout * 9 * L[l].Cin;
        unsigned g = (unsigned)((total + 255) / 256);
        if (L[l].kind == 0)
            expand_lift_k<<<g, 256, 0, stream>>>(w, Wp[l], L[l].Of, L[l].If);
        else if (L[l].kind == 1)
            expand_reg_k<<<g, 256, 0, stream>>>(w, Wp[l], L[l].Of, L[l].If);
        else
            expand_proj_k<<<g, 256, 0, stream>>>(w, Wp[l], L[l].Of, L[l].If);
        int rep = (L[l].kind == 2) ? 1 : 4;
        expand_bias_k<<<(L[l].Cout + 255) / 256, 256, 0, stream>>>(bsrc, Bp[l], L[l].Cout, rep);
    }

    // -------- helpers --------
    auto conv = [&](int l, const bf16_t* in, bf16_t* out, int H, int W) {
        launch_conv(in, Wp[l], Bp[l], out, H, W, L[l].Cin, L[l].Cout, stream);
    };
    auto pool = [&](const bf16_t* in, bf16_t* out, int C, int Ho, int Wo) {
        size_t total = (size_t)Bn * Ho * Wo * C;
        pool2_k<<<(unsigned)((total + 255) / 256), 256, 0, stream>>>(in, out, Bn, C, Ho, Wo);
    };
    auto up2 = [&](const bf16_t* in, bf16_t* out, int C, int Hi, int Wi, int CT) {
        size_t total = (size_t)Bn * (2 * Hi) * (2 * Wi) * C;
        up2_k<<<(unsigned)((total + 255) / 256), 256, 0, stream>>>(in, out, Bn, C, Hi, Wi, CT);
    };
    auto cpy = [&](const bf16_t* in, bf16_t* out, int C, int H, int W, int CT, int CO) {
        size_t total = (size_t)Bn * H * W * C;
        copyoff_k<<<(unsigned)((total + 255) / 256), 256, 0, stream>>>(in, out, Bn, C, H, W, CT, CO);
    };

    // -------- network --------
    const float* x = (const float*)d_in[0];
    {
        size_t total = (size_t)Bn * 256 * 256 * 32;
        cvtin_k<<<(unsigned)((total + 255) / 256), 256, 0, stream>>>(x, X, Bn, 2, 256, 256, 32);
    }

    conv(0, X,  A,  256, 256);            // d1c1: 32 -> 64
    conv(1, A,  S1, 256, 256);            // d1c2: 64 -> 64 (skip d1)
    pool(S1, A, 64, 128, 128);
    conv(2, A,  Bb, 128, 128);            // d2c1: 64 -> 128
    conv(3, Bb, S2, 128, 128);            // d2c2 (skip d2)
    pool(S2, A, 128, 64, 64);
    conv(4, A,  Bb, 64, 64);              // d3c1: 128 -> 256
    conv(5, Bb, S3, 64, 64);              // d3c2 (skip d3)
    pool(S3, A, 256, 32, 32);
    conv(6, A,  Bb, 32, 32);              // d4c1: 256 -> 512
    conv(7, Bb, S4, 32, 32);              // d4c2 (skip d4)
    pool(S4, A, 512, 16, 16);
    conv(8, A,  Bb, 16, 16);              // bn1: 512 -> 1024
    conv(9, Bb, A,  16, 16);              // bn2: 1024 -> 1024

    up2(A, Bb, 1024, 16, 16, 1536);       // concat1 = [up(b), d4]
    cpy(S4, Bb, 512, 32, 32, 1536, 1024);
    conv(10, Bb, A,  32, 32);             // u1c1: 1536 -> 512
    conv(11, A,  Bb, 32, 32);             // u1c2

    up2(Bb, A, 512, 32, 32, 768);         // concat2 = [up(u1), d3]
    cpy(S3, A, 256, 64, 64, 768, 512);
    conv(12, A,  Bb, 64, 64);             // u2c1: 768 -> 256
    conv(13, Bb, A,  64, 64);             // u2c2

    up2(A, Bb, 256, 64, 64, 384);         // concat3 = [up(u2), d2]
    cpy(S2, Bb, 128, 128, 128, 384, 256);
    conv(14, Bb, A,  128, 128);           // u3c1: 384 -> 128
    conv(15, A,  Bb, 128, 128);           // u3c2

    up2(Bb, A, 128, 128, 128, 192);       // concat4 = [up(u3), d1]
    cpy(S1, A, 64, 256, 256, 192, 128);
    conv(16, A,  Bb, 256, 256);           // u4c1: 192 -> 64
    conv(17, Bb, A,  256, 256);           // u4c2 (proj): 64 -> 16

    {
        const float* ow = (const float*)d_in[37];
        const float* ob = (const float*)d_in[38];
        size_t total = (size_t)Bn * 256 * 256;
        out1x1_k<<<(unsigned)((total + 255) / 256), 256, 0, stream>>>(
            A, ow, ob, (float*)d_out, Bn, 256, 256, 16);
    }
}
